// MultiHeadAttention_44650480009682
// MI455X (gfx1250) — matrix operand order repaired and checked by reading, not compiled, not run on hardware
//
#include <hip/hip_runtime.h>
#include <math.h>

// ---------------------------------------------------------------------------
// MI455X (gfx1250) fused MHA forward, bf16 WMMA + async LDS staging.
//   Stage 0: one-time convert  x -> bf16,  W_qkv/W_out -> transposed bf16 [N,K]
//   Stage 1: QKV GEMM (bf16 in / bf16 out) -> Q,K [bh][S][64], V^T [bh][64][S]
//   Stage 2: flash attention (v_wmma_f32_16x16x32_bf16 for QK^T and PV)
//   Stage 3: out-proj GEMM (bf16 in / fp32 out + bias) -> d_out
// All tile staging uses GLOBAL_LOAD_ASYNC_TO_LDS_B128 (ASYNCcnt) -- no VGPR
// round-trip; fp32<->bf16 conversion is paid exactly once per element.
// ---------------------------------------------------------------------------

typedef __bf16 bf16;
typedef __attribute__((ext_vector_type(16))) __bf16 v16bf;
typedef __attribute__((ext_vector_type(8)))  __bf16 v8bf;
typedef __attribute__((ext_vector_type(4)))  __bf16 v4bf;
typedef __attribute__((ext_vector_type(8)))  float  v8f;

union FragBF {
  v16bf v;
  v8bf  h[2];
};

__device__ __forceinline__ bf16 f2bf(float x) { return (bf16)x; }

__device__ __forceinline__ v8f wmma_bf16(const v16bf& a, const v16bf& b, const v8f& c) {
  return __builtin_amdgcn_wmma_f32_16x16x32_bf16(false, a, false, b, (short)0, c,
                                                 false, false);
}

// Async 16-byte global->LDS copy (gfx1250). VDST = LDS offset, VADDR = 64-bit
// global address; LDS address is the LDS base plus VDST (ISA 15.18.3). Tracked on ASYNCcnt.
__device__ __forceinline__ void async_copy_b128(void* lds_dst, const void* gsrc) {
  unsigned lds = (unsigned)(size_t)lds_dst;  // low 32 bits of generic ptr = LDS offset
  asm volatile("global_load_async_to_lds_b128 %0, %1, off"
               :
               : "v"(lds), "v"(gsrc)
               : "memory");
}
__device__ __forceinline__ void wait_async0() {
  asm volatile("s_wait_asynccnt 0x0" ::: "memory");
}

// ---------------------------------------------------------------------------
// Stage 0a: straight fp32 -> bf16 copy (4 elems/thread, b64 stores).
// ---------------------------------------------------------------------------
__global__ __launch_bounds__(256) void cvt_copy_kernel(const float* __restrict__ src,
                                                       bf16* __restrict__ dst) {
  int idx = blockIdx.x * 256 + threadIdx.x;
  const float4 v = reinterpret_cast<const float4*>(src)[idx];
  v4bf o = {f2bf(v.x), f2bf(v.y), f2bf(v.z), f2bf(v.w)};
  reinterpret_cast<v4bf*>(dst)[idx] = o;
}

// ---------------------------------------------------------------------------
// Stage 0b: fp32 [K,N] -> bf16 [N,K] transpose (coalesced reads).
// ---------------------------------------------------------------------------
__global__ __launch_bounds__(256) void cvt_transpose_kernel(
    const float* __restrict__ src, bf16* __restrict__ dst, int K, int N) {
  int idx = blockIdx.x * 256 + threadIdx.x;  // over K*N/4
  int nq = N >> 2;
  int k  = idx / nq;
  int n4 = (idx - k * nq) * 4;
  const float4 v =
      *reinterpret_cast<const float4*>(src + (size_t)k * N + n4);
  dst[(size_t)(n4 + 0) * K + k] = f2bf(v.x);
  dst[(size_t)(n4 + 1) * K + k] = f2bf(v.y);
  dst[(size_t)(n4 + 2) * K + k] = f2bf(v.z);
  dst[(size_t)(n4 + 3) * K + k] = f2bf(v.w);
}

// ---------------------------------------------------------------------------
// Tiled GEMM: C[M,N] = A[M,K] @ Bt[N,K]^T + bias[N]
// A, Bt already bf16; workgroup tile 128x128, 8 waves x (64x32), K-step 32.
// MODE 0: fp32 row-major store to C (out projection).
// MODE 1: bf16 scatter: Q/K -> [bh][s][64], V -> transposed [bh][64][s].
// ---------------------------------------------------------------------------
template <int MODE>
__global__ __launch_bounds__(256) void gemm_kernel(
    const bf16* __restrict__ A, const bf16* __restrict__ Bt,
    const float* __restrict__ bias, float* __restrict__ C,
    bf16* __restrict__ Qb, bf16* __restrict__ Kb, bf16* __restrict__ Vt,
    int M, int N, int K) {
  __shared__ __align__(16) bf16 sA[128][40];  // [m][k], 80B stride (16B aligned)
  __shared__ __align__(16) bf16 sB[128][40];  // [n][k]

  const int tid  = threadIdx.x;
  const int wave = tid >> 5;
  const int lane = tid & 31;
  const int half = lane >> 4;
  const int r16  = lane & 15;
  const int m0 = blockIdx.y * 128;
  const int n0 = blockIdx.x * 128;
  const int waveM = (wave & 1) * 64;
  const int waveN = (wave >> 1) * 32;

  v8f acc[4][2] = {};

  for (int k0 = 0; k0 < K; k0 += 32) {
    // Stage A and Bt tiles: 128x32 bf16 = 512 16B chunks each, async to LDS.
#pragma unroll
    for (int it = 0; it < 2; ++it) {
      int idx = tid + it * 256;  // 128 rows * 4 chunks
      int row = idx >> 2;
      int c8  = (idx & 3) * 8;
      async_copy_b128(&sA[row][c8], A + (size_t)(m0 + row) * K + k0 + c8);
      async_copy_b128(&sB[row][c8], Bt + (size_t)(n0 + row) * K + k0 + c8);
    }
    wait_async0();
    __syncthreads();

    // Fragment loads: two ds_load_b128 each (ISA 16-bit operand layouts).
    FragBF a[4], b[2];
#pragma unroll
    for (int mt = 0; mt < 4; ++mt) {
      const bf16* p = &sA[waveM + mt * 16 + r16][half * 8];
      a[mt].h[0] = *reinterpret_cast<const v8bf*>(p);       // k = 8*half + 0..7
      a[mt].h[1] = *reinterpret_cast<const v8bf*>(p + 16);  // k = 16+8*half+0..7
    }
#pragma unroll
    for (int nt = 0; nt < 2; ++nt) {
      const bf16* p = &sB[waveN + nt * 16 + r16][half * 8];
      b[nt].h[0] = *reinterpret_cast<const v8bf*>(p);       // k = 8*half + 0..7
      b[nt].h[1] = *reinterpret_cast<const v8bf*>(p + 16);  // k = 16+8*half+0..7
    }
#pragma unroll
    for (int mt = 0; mt < 4; ++mt)
#pragma unroll
      for (int nt = 0; nt < 2; ++nt)
        acc[mt][nt] = wmma_bf16(a[mt].v, b[nt].v, acc[mt][nt]);
    __syncthreads();
  }

  // Epilogue: C/D layout: lane r16 = col, VGPR j = row (j + 8*half).
#pragma unroll
  for (int mt = 0; mt < 4; ++mt) {
#pragma unroll
    for (int nt = 0; nt < 2; ++nt) {
#pragma unroll
      for (int j = 0; j < 8; ++j) {
        int m = m0 + waveM + mt * 16 + half * 8 + j;
        int n = n0 + waveN + nt * 16 + r16;
        float val = acc[mt][nt][j] + bias[n];
        if (MODE == 0) {
          C[(size_t)m * N + n] = val;
        } else {
          // n = h*192 + t*64 + d ; m = b*2048 + s
          int h = n / 192, rem = n - h * 192;
          int t = rem >> 6, d = rem & 63;
          int bb = m >> 11, s = m & 2047;
          int bh = bb * 16 + h;
          if (t == 0)
            Qb[((size_t)bh * 2048 + s) * 64 + d] = f2bf(val);
          else if (t == 1)
            Kb[((size_t)bh * 2048 + s) * 64 + d] = f2bf(val);
          else
            Vt[((size_t)bh * 64 + d) * 2048 + s] = f2bf(val);  // pre-transposed
        }
      }
    }
  }
}

// ---------------------------------------------------------------------------
// Flash attention: grid (B*H=32, S/128=16), 256 threads (8 waves).
// Q,K: bf16 [bh][s][64]; V: bf16 transposed [bh][64][s]; ctx: bf16 [B*S][1024].
// ---------------------------------------------------------------------------
__global__ __launch_bounds__(256) void attn_kernel(
    const bf16* __restrict__ Q, const bf16* __restrict__ K,
    const bf16* __restrict__ Vt, bf16* __restrict__ ctx) {
  constexpr int S = 2048, HD = 64;
  __shared__ __align__(16) bf16 sQ[128][72];    // [q][hd]
  __shared__ __align__(16) bf16 sK[64][72];     // [key][hd]  (B-frag for QK^T)
  __shared__ __align__(16) bf16 sVt[64][72];    // [hd][key]  (B-frag for PV)
  __shared__ __align__(16) bf16 sP[8][16][72];  // per-wave probs [q][key]

  const int tid  = threadIdx.x;
  const int wave = tid >> 5;
  const int lane = tid & 31;
  const int half = lane >> 4;
  const int r16  = lane & 15;
  const int bh = blockIdx.x;
  const int q0 = blockIdx.y * 128;
  const int bb = bh >> 4, hh = bh & 15;
  const int wq = wave * 16;

  const bf16* Qp  = Q + (size_t)bh * S * HD;
  const bf16* Kp  = K + (size_t)bh * S * HD;
  const bf16* Vtp = Vt + (size_t)bh * HD * S;

  // Stage Q tile 128x64 bf16 async (completion covered by first in-loop wait).
#pragma unroll
  for (int it = 0; it < 4; ++it) {
    int idx = tid + it * 256;  // 128 rows * 8 chunks
    int row = idx >> 3;
    int c8  = (idx & 7) * 8;
    async_copy_b128(&sQ[row][c8], Qp + (size_t)(q0 + row) * HD + c8);
  }

  float mrow[8], lrow[8];
  v8f oacc[4] = {};
#pragma unroll
  for (int j = 0; j < 8; ++j) { mrow[j] = -__builtin_inff(); lrow[j] = 0.0f; }

  const int kend = q0 + 128;
  for (int kb = 0; kb < kend; kb += 64) {
    // Stage K [key][hd] and V^T [hd][key]: 512 chunks each, async to LDS.
#pragma unroll
    for (int it = 0; it < 2; ++it) {
      int idx = tid + it * 256;  // 64 rows * 8 chunks
      int row = idx >> 3;
      int c8  = (idx & 7) * 8;
      async_copy_b128(&sK[row][c8], Kp + (size_t)(kb + row) * HD + c8);
      async_copy_b128(&sVt[row][c8], Vtp + (size_t)row * S + kb + c8);
    }
    wait_async0();
    __syncthreads();

    // Scores: S = Q(16x64) @ K^T(64x64) -> 4 accumulators of 16x16.
    FragBF qa[2];
#pragma unroll
    for (int ks = 0; ks < 2; ++ks) {
      const bf16* p = &sQ[wq + r16][ks * 32 + half * 8];
      qa[ks].h[0] = *reinterpret_cast<const v8bf*>(p);
      qa[ks].h[1] = *reinterpret_cast<const v8bf*>(p + 16);
    }
    v8f sc[4] = {};
#pragma unroll
    for (int nt = 0; nt < 4; ++nt) {
#pragma unroll
      for (int ks = 0; ks < 2; ++ks) {
        FragBF kf;
        const bf16* p = &sK[nt * 16 + r16][ks * 32 + half * 8];
        kf.h[0] = *reinterpret_cast<const v8bf*>(p);
        kf.h[1] = *reinterpret_cast<const v8bf*>(p + 16);
        sc[nt] = wmma_bf16(qa[ks].v, kf.v, sc[nt]);
      }
    }

    // Online softmax (per row: 16-lane butterfly reductions, wave32).
#pragma unroll
    for (int j = 0; j < 8; ++j) {
      const int rr = q0 + wq + half * 8 + j;  // global query row
      float vals[4];
#pragma unroll
      for (int nt = 0; nt < 4; ++nt) {
        int cc = kb + nt * 16 + r16;           // global key col
        float sv = sc[nt][j] * 0.125f;         // 1/sqrt(64)
        vals[nt] = (cc > rr) ? -1e9f : sv;     // causal mask == clamp(-inf)
      }
      float mx = fmaxf(fmaxf(vals[0], vals[1]), fmaxf(vals[2], vals[3]));
#pragma unroll
      for (int off = 1; off < 16; off <<= 1)
        mx = fmaxf(mx, __shfl_xor(mx, off, 32));
      float newm  = fmaxf(mrow[j], mx);
      float alpha = __expf(mrow[j] - newm);
      float psum = 0.0f;
#pragma unroll
      for (int nt = 0; nt < 4; ++nt) {
        float pv = __expf(vals[nt] - newm);
        psum += pv;
        sP[wave][half * 8 + j][nt * 16 + r16] = f2bf(pv);
      }
#pragma unroll
      for (int off = 1; off < 16; off <<= 1)
        psum += __shfl_xor(psum, off, 32);
      lrow[j] = lrow[j] * alpha + psum;
      mrow[j] = newm;
#pragma unroll
      for (int nt = 0; nt < 4; ++nt) oacc[nt][j] *= alpha;
    }
    __syncthreads();  // sP visible across lanes

    // O += P(16x64) @ V(64x64).
    FragBF pa[2];
#pragma unroll
    for (int ks = 0; ks < 2; ++ks) {
      const bf16* p = &sP[wave][r16][ks * 32 + half * 8];
      pa[ks].h[0] = *reinterpret_cast<const v8bf*>(p);
      pa[ks].h[1] = *reinterpret_cast<const v8bf*>(p + 16);
    }
#pragma unroll
    for (int nt = 0; nt < 4; ++nt) {
#pragma unroll
      for (int ks = 0; ks < 2; ++ks) {
        FragBF vf;
        const bf16* p = &sVt[nt * 16 + r16][ks * 32 + half * 8];
        vf.h[0] = *reinterpret_cast<const v8bf*>(p);
        vf.h[1] = *reinterpret_cast<const v8bf*>(p + 16);
        oacc[nt] = wmma_bf16(pa[ks].v, vf.v, oacc[nt]);
      }
    }
    __syncthreads();  // before next tile overwrites sK/sVt/sP
  }

  // Normalize; write ctx[b, s, h*64+d] (bf16) for the out-proj GEMM.
#pragma unroll
  for (int nt = 0; nt < 4; ++nt) {
#pragma unroll
    for (int j = 0; j < 8; ++j) {
      int qg  = q0 + wq + half * 8 + j;
      int col = hh * 64 + nt * 16 + r16;
      ctx[((size_t)(bb * 2048 + qg)) * 1024 + col] = f2bf(oacc[nt][j] / lrow[j]);
    }
  }
}

// ---------------------------------------------------------------------------
extern "C" void kernel_launch(void* const* d_in, const int* in_sizes, int n_in,
                              void* d_out, int out_size, void* d_ws, size_t ws_size,
                              hipStream_t stream) {
  const float* x     = (const float*)d_in[0];
  const float* W_qkv = (const float*)d_in[1];
  const float* b_qkv = (const float*)d_in[2];
  const float* W_out = (const float*)d_in[3];
  const float* b_out = (const float*)d_in[4];
  float* out = (float*)d_out;

  // Workspace layout (bf16 elements).
  const size_t XB   = (size_t)4096 * 1024;     // x       : 8 MB
  const size_t WQT  = (size_t)3072 * 1024;     // W_qkv^T : 6 MB
  const size_t WOT  = (size_t)1024 * 1024;     // W_out^T : 2 MB
  const size_t HBUF = (size_t)32 * 2048 * 64;  // Q/K/V^T : 8 MB each
  bf16* xb    = (bf16*)d_ws;
  bf16* Wqkvt = xb + XB;
  bf16* Woutt = Wqkvt + WQT;
  bf16* Qb    = Woutt + WOT;
  bf16* Kb    = Qb + HBUF;
  bf16* Vt    = Kb + HBUF;
  bf16* ctx   = Vt + HBUF;

  // Stage 0: one-time conversions.
  cvt_copy_kernel<<<(4096 * 1024 / 4) / 256, 256, 0, stream>>>(x, xb);
  cvt_transpose_kernel<<<(1024 * 3072 / 4) / 256, 256, 0, stream>>>(W_qkv, Wqkvt,
                                                                    1024, 3072);
  cvt_transpose_kernel<<<(1024 * 1024 / 4) / 256, 256, 0, stream>>>(W_out, Woutt,
                                                                    1024, 1024);

  // Stage 1: QKV projection (M=4096, N=3072, K=1024) -> Q/K/V^T bf16.
  gemm_kernel<1><<<dim3(3072 / 128, 4096 / 128), 256, 0, stream>>>(
      xb, Wqkvt, b_qkv, nullptr, Qb, Kb, Vt, 4096, 3072, 1024);

  // Stage 2: causal flash attention.
  attn_kernel<<<dim3(32, 16), 256, 0, stream>>>(Qb, Kb, Vt, ctx);

  // Stage 3: output projection (M=4096, N=1024, K=1024) -> fp32 d_out.
  gemm_kernel<0><<<dim3(1024 / 128, 4096 / 128), 256, 0, stream>>>(
      ctx, Woutt, b_out, out, nullptr, nullptr, nullptr, 4096, 1024, 1024);
}
